// LSTM_8504035246515
// MI455X (gfx1250) — matrix operand order repaired and checked by reading, not compiled, not run on hardware
//
#include <hip/hip_runtime.h>

// ---------------------------------------------------------------------------
// LSTM forward for MI455X (gfx1250, wave32, WMMA bf16)
// B=64, T=256, I=512, H=1024, 4H=4096
// ---------------------------------------------------------------------------

typedef __attribute__((ext_vector_type(16))) __bf16 v16bf;
typedef __attribute__((ext_vector_type(8)))  __bf16 v8bf;
typedef __attribute__((ext_vector_type(8)))  float  v8f;

#define LSTM_B   64
#define LSTM_T   256
#define LSTM_I   512
#define LSTM_H   1024
#define LSTM_G   4096   // 4*H
#define LSTM_M   16384  // B*T

__device__ __forceinline__ float sigmoid_f(float x) {
    return 1.0f / (1.0f + __expf(-x));
}
__device__ __forceinline__ float tanh_f(float x) {
    return 2.0f / (1.0f + __expf(-2.0f * x)) - 1.0f;
}

// ---------------------------------------------------------------------------
// Zero h (both ping-pong buffers) and c. 65536 threads.
// ---------------------------------------------------------------------------
__global__ void lstm_init_kernel(float* __restrict__ c,
                                 __bf16* __restrict__ h0,
                                 __bf16* __restrict__ h1) {
    int i = blockIdx.x * blockDim.x + threadIdx.x;
    c[i]  = 0.0f;
    h0[i] = (__bf16)0.0f;
    h1[i] = (__bf16)0.0f;
}

// ---------------------------------------------------------------------------
// f32 -> bf16 elementwise convert
// ---------------------------------------------------------------------------
__global__ void cvt_bf16_kernel(const float* __restrict__ s,
                                __bf16* __restrict__ d, int n) {
    int i = blockIdx.x * blockDim.x + threadIdx.x;
    if (i < n) d[i] = (__bf16)s[i];
}

// ---------------------------------------------------------------------------
// Pack a row-major [K, N] f32 weight matrix into bf16 WMMA B-fragment order.
// Tile = 32(K) x 16(N). Packed element address:
//   ((k0/32)*(N/16) + n0/16)*512 + lane*16 + e
// holds W[k0 + (e&7) + 8*(lane>=16) + 16*(e>=8)][n0 + (lane&15)]
// ---------------------------------------------------------------------------
__global__ void pack_b_kernel(const float* __restrict__ W,
                              __bf16* __restrict__ dst, int K, int N) {
    int idx = blockIdx.x * blockDim.x + threadIdx.x;
    if (idx >= K * N) return;
    int e    = idx & 15;
    int l    = (idx >> 4) & 31;
    int tile = idx >> 9;
    int ntn  = N >> 4;
    int n0   = (tile % ntn) << 4;
    int k0   = (tile / ntn) << 5;
    int k    = k0 + (e & 7) + ((l >> 4) << 3) + ((e >> 3) << 4);
    int n    = n0 + (l & 15);
    dst[idx] = (__bf16)W[(size_t)k * N + n];
}

// ---------------------------------------------------------------------------
// x_proj = x @ Wx + b  :  [16384,512] x [512,4096] -> f32 [16384,4096]
// Block = 256 thr = 8 waves. Wave computes 16 rows x 64 cols (4 WMMA tiles).
// grid = (8, 1024)
// ---------------------------------------------------------------------------
__global__ __launch_bounds__(256)
void gemm_xproj_kernel(const __bf16* __restrict__ Xb,
                       const __bf16* __restrict__ Wxp,
                       const float*  __restrict__ bias,
                       float* __restrict__ xproj) {
    const int lane   = threadIdx.x & 31;
    const int wave   = threadIdx.x >> 5;
    const int m0     = blockIdx.y << 4;
    const int n0     = (blockIdx.x << 9) + (wave << 6);
    const int laneHi = lane >> 4;
    const int lmod   = lane & 15;

    v8f acc[4] = {};
    const __bf16* arow = Xb + (size_t)(m0 + lmod) * LSTM_I;

    for (int k0 = 0; k0 < LSTM_I; k0 += 32) {
        v8bf alo = *(const v8bf*)(arow + k0 + laneHi * 8);
        v8bf ahi = *(const v8bf*)(arow + k0 + 16 + laneHi * 8);
        v16bf afrag;
#pragma unroll
        for (int e = 0; e < 8; ++e) { afrag[e] = alo[e]; afrag[e + 8] = ahi[e]; }

        const __bf16* bbase =
            Wxp + ((size_t)(k0 >> 5) * (LSTM_G >> 4) + (n0 >> 4)) * 512 + lane * 16;
#pragma unroll
        for (int j = 0; j < 4; ++j) {
            v16bf bfrag = *(const v16bf*)(bbase + (size_t)j * 512);
            acc[j] = __builtin_amdgcn_wmma_f32_16x16x32_bf16(
                false, afrag, false, bfrag, (short)0, acc[j], false, false);
        }
    }

#pragma unroll
    for (int j = 0; j < 4; ++j) {
        int col  = n0 + j * 16 + lmod;
        float bv = bias[col];
#pragma unroll
        for (int r = 0; r < 8; ++r) {
            int row = m0 + r + laneHi * 8;
            xproj[(size_t)row * LSTM_G + col] = acc[j][r] + bv;
        }
    }
}

// ---------------------------------------------------------------------------
// One LSTM timestep, fused, with split-K for 4x occupancy on the critical
// path:
//   - block (256 thr = 8 waves) owns TWO (16-batch x 16-hidden) tiles
//   - per tile, 4 waves each compute a K=256 chunk of all 4 gate WMMA tiles
//   - f32 partials reduced through LDS (stride-18 pad: conflict-free)
//   - LSTM cell applied on reduced gates; h written f32 (out) + bf16 (next t)
// grid = 128 blocks. c is element-exclusive (no race); h double-buffered.
// ---------------------------------------------------------------------------
__global__ __launch_bounds__(256)
void lstm_step_kernel(const __bf16* __restrict__ Whp,   // packed [1024 -> 4096]
                      const float*  __restrict__ xproj, // [16384,4096], row b*T+t
                      const __bf16* __restrict__ h_in,  // [64,1024] bf16
                      __bf16* __restrict__ h_out,       // [64,1024] bf16
                      float* __restrict__ cst,          // [64,1024] f32
                      float* __restrict__ out,          // [64,256,1024] f32
                      int t) {
    // [pair][kwave][gate][m][n(+pad)] : 2*4*4*16*18*4 = 36,864 B
    __shared__ float red[2][4][4][16][18];

    const int lane   = threadIdx.x & 31;
    const int wave   = threadIdx.x >> 5;   // 0..7
    const int pair   = wave >> 2;          // 0..1 : which tile this wave helps
    const int kw     = wave & 3;           // 0..3 : K chunk id
    const int tp     = (blockIdx.x << 1) + pair;  // global tile id 0..255
    const int m0     = (tp >> 6) << 4;     // batch tile base
    const int htile  = tp & 63;
    const int h0     = htile << 4;         // hidden tile base
    const int laneHi = lane >> 4;
    const int lmod   = lane & 15;

    v8f acc[4] = {};  // i, f, g, o partial accumulators for K chunk
    const __bf16* arow = h_in + (size_t)(m0 + lmod) * LSTM_H;
    const int kbeg = kw << 8;              // kw * 256

    for (int kk = 0; kk < 256; kk += 32) {
        const int k0 = kbeg + kk;
        v8bf alo = *(const v8bf*)(arow + k0 + laneHi * 8);
        v8bf ahi = *(const v8bf*)(arow + k0 + 16 + laneHi * 8);
        v16bf afrag;
#pragma unroll
        for (int e = 0; e < 8; ++e) { afrag[e] = alo[e]; afrag[e + 8] = ahi[e]; }

        const size_t kb = (size_t)(k0 >> 5) * (LSTM_G >> 4);
#pragma unroll
        for (int g = 0; g < 4; ++g) {  // gate tiles at n = g*1024 + h0
            v16bf bfrag = *(const v16bf*)(Whp + (kb + g * 64 + htile) * 512 + lane * 16);
            acc[g] = __builtin_amdgcn_wmma_f32_16x16x32_bf16(
                false, afrag, false, bfrag, (short)0, acc[g], false, false);
        }
    }

    // stash partials: C/D elem r -> row r + 8*laneHi, col lmod
#pragma unroll
    for (int g = 0; g < 4; ++g)
#pragma unroll
        for (int r = 0; r < 8; ++r)
            red[pair][kw][g][r + 8 * laneHi][lmod] = acc[g][r];

    __syncthreads();

    // cell: 4 waves (128 lanes) per tile, 2 elements per lane = 256 elems
    const int lig = (kw << 5) + lane;  // 0..127 within this tile's wave group
#pragma unroll
    for (int q = 0; q < 2; ++q) {
        int elem = lig + (q << 7);     // 0..255
        int mi   = elem >> 4;          // 0..15
        int ni   = elem & 15;
        float gi = red[pair][0][0][mi][ni] + red[pair][1][0][mi][ni] +
                   red[pair][2][0][mi][ni] + red[pair][3][0][mi][ni];
        float gf = red[pair][0][1][mi][ni] + red[pair][1][1][mi][ni] +
                   red[pair][2][1][mi][ni] + red[pair][3][1][mi][ni];
        float gg = red[pair][0][2][mi][ni] + red[pair][1][2][mi][ni] +
                   red[pair][2][2][mi][ni] + red[pair][3][2][mi][ni];
        float go = red[pair][0][3][mi][ni] + red[pair][1][3][mi][ni] +
                   red[pair][2][3][mi][ni] + red[pair][3][3][mi][ni];

        int b   = m0 + mi;
        int col = h0 + ni;
        size_t xrow = (size_t)(b * LSTM_T + t) * LSTM_G;
        gi += xproj[xrow + 0 * LSTM_H + col];
        gf += xproj[xrow + 1 * LSTM_H + col];
        gg += xproj[xrow + 2 * LSTM_H + col];
        go += xproj[xrow + 3 * LSTM_H + col];

        float iv = sigmoid_f(gi);
        float fv = sigmoid_f(gf);
        float gv = tanh_f(gg);
        float ov = sigmoid_f(go);

        size_t cidx = (size_t)b * LSTM_H + col;
        float cv = fv * cst[cidx] + iv * gv;
        cst[cidx] = cv;
        float hv = ov * tanh_f(cv);
        out[(size_t)(b * LSTM_T + t) * LSTM_H + col] = hv;
        h_out[cidx] = (__bf16)hv;
    }
}

// ---------------------------------------------------------------------------
// Launch
// ---------------------------------------------------------------------------
extern "C" void kernel_launch(void* const* d_in, const int* in_sizes, int n_in,
                              void* d_out, int out_size, void* d_ws, size_t ws_size,
                              hipStream_t stream) {
    const float* x  = (const float*)d_in[0];  // [64,256,512]
    const float* Wx = (const float*)d_in[1];  // [512,4096]
    const float* Wh = (const float*)d_in[2];  // [1024,4096]
    const float* b  = (const float*)d_in[3];  // [4096]
    float* out = (float*)d_out;               // [64,256,1024]

    char* ws = (char*)d_ws;
    // workspace layout (all offsets 32B aligned)
    __bf16* x_bf  = (__bf16*)(ws + 0);            // 16384*512*2   = 16,777,216
    __bf16* Wxp   = (__bf16*)(ws + 16777216);     // 512*4096*2    =  4,194,304
    __bf16* Whp   = (__bf16*)(ws + 20971520);     // 1024*4096*2   =  8,388,608
    __bf16* hbuf0 = (__bf16*)(ws + 29360128);     // 64*1024*2     =    131,072
    __bf16* hbuf1 = (__bf16*)(ws + 29491200);     // 64*1024*2     =    131,072
    float*  cst   = (float*) (ws + 29622272);     // 64*1024*4     =    262,144
    float*  xproj = (float*) (ws + 29884416);     // 16384*4096*4  = 268,435,456

    // 0) zero recurrent state
    lstm_init_kernel<<<LSTM_B * LSTM_H / 256, 256, 0, stream>>>(cst, hbuf0, hbuf1);

    // 1) convert x to bf16
    {
        int n = LSTM_M * LSTM_I;  // 8,388,608
        cvt_bf16_kernel<<<(n + 255) / 256, 256, 0, stream>>>(x, x_bf, n);
    }

    // 2) pack weights into WMMA B-fragment order
    pack_b_kernel<<<(LSTM_I * LSTM_G + 255) / 256, 256, 0, stream>>>(Wx, Wxp, LSTM_I, LSTM_G);
    pack_b_kernel<<<(LSTM_H * LSTM_G + 255) / 256, 256, 0, stream>>>(Wh, Whp, LSTM_H, LSTM_G);

    // 3) x_proj = x @ Wx + b
    {
        dim3 grid(LSTM_G / 512, LSTM_M / 16);  // (8, 1024)
        gemm_xproj_kernel<<<grid, 256, 0, stream>>>(x_bf, Wxp, b, xproj);
    }

    // 4) sequential recurrence, one fused split-K kernel per timestep
    for (int t = 0; t < LSTM_T; ++t) {
        const __bf16* hin = (t & 1) ? hbuf1 : hbuf0;
        __bf16* hout      = (t & 1) ? hbuf0 : hbuf1;
        lstm_step_kernel<<<128, 256, 0, stream>>>(Whp, xproj, hin, hout, cst, out, t);
    }
}
